// GeomAttention_6073083756585
// MI455X (gfx1250) — matrix operand order repaired and checked by reading, not compiled, not run on hardware
//
#include <hip/hip_runtime.h>

typedef __attribute__((ext_vector_type(16))) _Float16 v16h;
typedef __attribute__((ext_vector_type(8)))  _Float16 v8h;
typedef __attribute__((ext_vector_type(8)))  float    v8f;

#define B_  2
#define L_  2048
#define H_  8
#define E_  32
#define D_  32

// Flash-style fused geometric attention.
// Block: 128 threads (4 waves). Each wave owns 32 query rows (two 16-row A tiles);
// block covers 128 rows. Loop over key/value chunks of 32, staged fp32->f16 in LDS.
// B operands (keys, transposed values) and key norms are shared across both A tiles.
__global__ __launch_bounds__(128) void geom_attn_kernel(
    const float* __restrict__ Q, const float* __restrict__ K,
    const float* __restrict__ V, float* __restrict__ O)
{
    __shared__ _Float16 kbuf[32 * 32];        // K chunk, row-major [key][e], f16
    __shared__ _Float16 vbufT[32 * 32];       // V chunk, transposed [d][s], f16
    __shared__ float    kn2s[32];             // key squared norms (fp32, exact)
    __shared__ _Float16 pbuf[4][2][16 * 32];  // per-wave, per-qtile P relayout scratch

    const int tid  = threadIdx.x;
    const int wave = tid >> 5;
    const int lane = tid & 31;
    const int hl   = lane & 15;               // half-lane index
    const int hi   = (lane >> 4) & 1;         // 0: lanes 0-15, 1: lanes 16-31

    const int bh = blockIdx.y;
    const int b  = bh / H_;
    const int h  = bh % H_;
    const int row0 = blockIdx.x * 128 + wave * 32;

    const float c = 0.0883883476483184f;      // 0.5 * (1/sqrt(E)) ; alpha = 0.5

    // ---- Load my two Q tiles as WMMA A operands (16x32 f16) + exact fp32 row norms.
    // A layout: lane<16: row=hl, k = {0..7, 16..23}; lane>=16: row=hl, k = {8..15, 24..31}
    const int eb = hi * 8;
    v16h aQ[2];
    float cqn2v[2][8];
    float mrow[2][8], lrow[2][8];
    v8f acc0[2], acc1[2];
#pragma unroll
    for (int qt = 0; qt < 2; ++qt) {
        const int qrow = row0 + qt * 16 + hl;
        const float* qp = Q + (((size_t)b * L_ + qrow) * H_ + h) * E_;
        float qpart = 0.f;
#pragma unroll
        for (int half = 0; half < 2; ++half) {
            const float* src = qp + eb + half * 16;
#pragma unroll
            for (int i = 0; i < 8; ++i) {
                float f = src[i];
                qpart += f * f;
                aQ[qt][half * 8 + i] = (_Float16)f;
            }
        }
        // combine the two e-halves (lane r and lane r^16 both hold row r)
        float qn2row = qpart + __shfl_xor(qpart, 16, 32);
        // distribute pre-scaled qn2 to C-layout rows: vgpr v <-> row M = v + 8*hi
#pragma unroll
        for (int v = 0; v < 8; ++v)
            cqn2v[qt][v] = c * __shfl(qn2row, v + hi * 8, 32);
#pragma unroll
        for (int v = 0; v < 8; ++v) { mrow[qt][v] = -1e30f; lrow[qt][v] = 0.f; }
        acc0[qt] = (v8f){};
        acc1[qt] = (v8f){};
    }

    for (int s0 = 0; s0 < L_; s0 += 32) {
        __syncthreads();  // protect kbuf/vbufT/kn2s reuse from previous iteration
        // ---- Cooperative staging (128 threads).
        {
            // K: 4 threads per key row, 8 elements each -> one packed ds_store_b128
            const int r  = tid >> 2;          // 0..31 key row
            const int c0 = (tid & 3) * 8;     // start e
            const float* kp = K + (((size_t)b * L_ + (s0 + r)) * H_ + h) * E_ + c0;
            float kf[8];
            v8h   kh;
            float p = 0.f;
#pragma unroll
            for (int i = 0; i < 8; ++i) {
                kf[i] = kp[i];
                p += kf[i] * kf[i];
                kh[i] = (_Float16)kf[i];
            }
            *(v8h*)&kbuf[r * 32 + c0] = kh;
            p += __shfl_xor(p, 1, 32);
            p += __shfl_xor(p, 2, 32);
            if ((tid & 3) == 0) kn2s[r] = p;

            // V: one d-column per lane (coalesced global loads), 8 s-values each,
            // packed into one contiguous ds_store_b128 of the transposed buffer.
            const int d    = tid & 31;
            const int sblk = (tid >> 5) * 8;
            const float* vp = V + (((size_t)b * L_ + (s0 + sblk)) * H_ + h) * D_ + d;
            v8h vh;
#pragma unroll
            for (int i = 0; i < 8; ++i)
                vh[i] = (_Float16)vp[(size_t)i * (H_ * D_)];
            *(v8h*)&vbufT[d * 32 + sblk] = vh;
        }
        __syncthreads();

        // ---- Shared B operands for this chunk.
        // QK^T B layout: lane<16: col N=hl (key), k=e {0..7, 16..23}; lane>=16: k=e {8..15, 24..31}
        v16h bK0, bK1, bV0, bV1;
        {
            const _Float16* k0p = &kbuf[(0  + hl) * 32 + hi * 8];
            const _Float16* k1p = &kbuf[(16 + hl) * 32 + hi * 8];
            const _Float16* v0p = &vbufT[(0  + hl) * 32 + hi * 8];
            const _Float16* v1p = &vbufT[(16 + hl) * 32 + hi * 8];
#pragma unroll
            for (int i = 0; i < 8; ++i) {
                bK0[i] = k0p[i]; bK1[i] = k1p[i];
                bV0[i] = v0p[i]; bV1[i] = v1p[i];
                bK0[8 + i] = k0p[16 + i]; bK1[8 + i] = k1p[16 + i];
                bV0[8 + i] = v0p[16 + i]; bV1[8 + i] = v1p[16 + i];
            }
        }
        const float kn2a = kn2s[hl];
        const float kn2b = kn2s[16 + hl];

#pragma unroll
        for (int qt = 0; qt < 2; ++qt) {
            v8f dot0 = {}, dot1 = {};
            dot0 = __builtin_amdgcn_wmma_f32_16x16x32_f16(false, aQ[qt], false, bK0, (short)0, dot0, false, false);
            dot1 = __builtin_amdgcn_wmma_f32_16x16x32_f16(false, aQ[qt], false, bK1, (short)0, dot1, false, false);

            // ---- scores + online softmax (row M = v + 8*hi spans a 16-lane group)
            // score = c*dot - relu(c*qn2*kn2 - (c*dot)*dot)
            float p0[8], p1[8];
#pragma unroll
            for (int v = 0; v < 8; ++v) {
                float d0 = dot0[v], d1 = dot1[v];
                float cd0 = c * d0, cd1 = c * d1;
                float sc0 = cd0 - fmaxf(cqn2v[qt][v] * kn2a - cd0 * d0, 0.f);
                float sc1 = cd1 - fmaxf(cqn2v[qt][v] * kn2b - cd1 * d1, 0.f);
                float mc = fmaxf(sc0, sc1);
#pragma unroll
                for (int m = 1; m <= 8; m <<= 1) mc = fmaxf(mc, __shfl_xor(mc, m, 32));
                float mnew = fmaxf(mrow[qt][v], mc);
                float corr = __expf(mrow[qt][v] - mnew);
                float e0 = __expf(sc0 - mnew);
                float e1 = __expf(sc1 - mnew);
                float rs = e0 + e1;
#pragma unroll
                for (int m = 1; m <= 8; m <<= 1) rs += __shfl_xor(rs, m, 32);
                lrow[qt][v] = lrow[qt][v] * corr + rs;
                mrow[qt][v] = mnew;
                acc0[qt][v] *= corr;
                acc1[qt][v] *= corr;
                p0[v] = e0;
                p1[v] = e1;
            }

            // ---- relayout P: C layout -> A layout via wave-private LDS scratch
            _Float16* pb = pbuf[wave][qt];
#pragma unroll
            for (int v = 0; v < 8; ++v) {
                const int M = v + hi * 8;
                pb[M * 32 + hl]      = (_Float16)p0[v];
                pb[M * 32 + 16 + hl] = (_Float16)p1[v];
            }
            // wave-internal LDS RAW fence (cheaper than a block barrier)
            asm volatile("s_wait_dscnt 0" ::: "memory");

            v16h aP;
            {
                const _Float16* src = &pb[hl * 32 + hi * 8];
#pragma unroll
                for (int i = 0; i < 8; ++i) { aP[i] = src[i]; aP[8 + i] = src[16 + i]; }
            }
            // ---- P * V (accumulate into running output)
            acc0[qt] = __builtin_amdgcn_wmma_f32_16x16x32_f16(false, aP, false, bV0, (short)0, acc0[qt], false, false);
            acc1[qt] = __builtin_amdgcn_wmma_f32_16x16x32_f16(false, aP, false, bV1, (short)0, acc1[qt], false, false);
        }
    }

    // ---- finalize: divide by softmax denominator, write fp32 output [B,L,H,D]
#pragma unroll
    for (int qt = 0; qt < 2; ++qt) {
#pragma unroll
        for (int v = 0; v < 8; ++v) {
            const int r = row0 + qt * 16 + v + hi * 8;
            const float inv = 1.0f / lrow[qt][v];
            float* op = O + (((size_t)b * L_ + r) * H_ + h) * D_;
            op[hl]      = acc0[qt][v] * inv;
            op[16 + hl] = acc1[qt][v] * inv;
        }
    }
}

extern "C" void kernel_launch(void* const* d_in, const int* in_sizes, int n_in,
                              void* d_out, int out_size, void* d_ws, size_t ws_size,
                              hipStream_t stream) {
    (void)in_sizes; (void)n_in; (void)out_size; (void)d_ws; (void)ws_size;
    const float* Q = (const float*)d_in[0];
    const float* K = (const float*)d_in[1];
    const float* V = (const float*)d_in[2];
    float* O = (float*)d_out;
    dim3 grid(L_ / 128, B_ * H_);
    dim3 block(128);
    geom_attn_kernel<<<grid, block, 0, stream>>>(Q, K, V, O);
}
